// DotProductAttention_79620103733324
// MI455X (gfx1250) — matrix operand order repaired and checked by reading, not compiled, not run on hardware
//
#include <hip/hip_runtime.h>

typedef __attribute__((ext_vector_type(16))) _Float16 v16h;
typedef __attribute__((ext_vector_type(8)))  float    v8f;

namespace {
constexpr int kB  = 2;
constexpr int kSQ = 2048;
constexpr int kSK = 2048;
constexpr int kNH = 32;
constexpr int kHN = 64;
constexpr int kQT = 128;   // query rows per workgroup (8 waves x 16)
constexpr int kKT = 64;    // kv rows per main-loop iteration
constexpr int kTiles = kSK / kKT;
// fold 1/sqrt(64) and log2(e) into Q so softmax runs in base-2
constexpr float kQScale = 0.125f * 1.44269504088896340736f;
}

__device__ __forceinline__ float fast_exp2(float x) {
    return __builtin_amdgcn_exp2f(x);   // v_exp_f32 (base-2 on AMD hardware)
}

__global__ __launch_bounds__(256, 1)
void fattn_fwd_f16wmma(const float* __restrict__ Q, const float* __restrict__ K,
                       const float* __restrict__ V, float* __restrict__ Out)
{
    // double-buffered K/V tiles: 2 x (8 KB + 8 KB) = 32 KB LDS
    __shared__ _Float16 sK[2][kKT][kHN];    // K tile, fp16
    __shared__ _Float16 sVt[2][kHN][kKT];   // V tile transposed

    const int tid  = threadIdx.x;
    const int lane = tid & 31;
    const int lm   = lane & 15;   // 0..15
    const int lh   = lane >> 4;   // 0..1 (lane half)
    const int wave = tid >> 5;

    const int bx    = blockIdx.x;
    const int qblk  = bx & (kSQ / kQT - 1);
    const int bh    = bx >> 4;
    const int head  = bh & (kNH - 1);
    const int batch = bh >> 5;

    const int q0 = qblk * kQT + wave * 16;

    // ---- Q^T as resident B-operand fragments: contiguous 16-float runs ----
    // B layout: N = lm (query row), Kdim(h) = ch*32 + lh*16 + e
    v16h qb[2];
    {
        const float* qrow =
            Q + (((size_t)batch * kSQ + (size_t)(q0 + lm)) * kNH + head) * kHN;
        #pragma unroll
        for (int ch = 0; ch < 2; ++ch) {
            #pragma unroll
            for (int e = 0; e < 16; ++e)
                qb[ch][e] = (_Float16)(qrow[ch * 32 + lh * 16 + e] * kQScale);
        }
    }

    // O accumulator in standard C layout: row q = r + 8*lh, col h = t*16 + lm
    v8f oacc[4];
    #pragma unroll
    for (int t = 0; t < 4; ++t) {
        #pragma unroll
        for (int r = 0; r < 8; ++r) oacc[t][r] = 0.0f;
    }
    // per-lane softmax state for query row q0 + lm (replicated across lh)
    float mreg = -1e30f, lreg = 0.0f;

    // ---- software-pipelined staging state ----
    const int    key  = tid >> 2;            // 0..63
    const int    hb   = (tid & 3) * 16;      // 0,16,32,48
    const size_t kvBase =
        (((size_t)batch * kSK) * kNH + head) * kHN + (size_t)key * (kNH * kHN) + hb;
    const size_t tileStride = (size_t)kKT * kNH * kHN;

    float kreg[16], vreg[16];

    auto load_tile = [&](int ti) {    // global -> registers (issue early)
        const float* kp = K + kvBase + (size_t)ti * tileStride;
        const float* vp = V + kvBase + (size_t)ti * tileStride;
        #pragma unroll
        for (int e = 0; e < 16; ++e) { kreg[e] = kp[e]; vreg[e] = vp[e]; }
        if (ti + 1 < kTiles) {        // keep L2 one more tile ahead
            __builtin_prefetch(kp + tileStride, 0, 1);
            __builtin_prefetch(vp + tileStride, 0, 1);
        }
    };
    auto store_tile = [&](int buf) {  // registers -> LDS (fp16; V transposed)
        #pragma unroll
        for (int e = 0; e < 16; ++e) {
            sK[buf][key][hb + e]  = (_Float16)kreg[e];
            sVt[buf][hb + e][key] = (_Float16)vreg[e];
        }
    };

    // prologue: fill buffer 0
    load_tile(0);
    store_tile(0);

    int cur = 0;
    for (int ti = 0; ti < kTiles; ++ti) {
        const bool more = (ti + 1 < kTiles);
        if (more) load_tile(ti + 1);   // overlap global latency with compute
        __syncthreads();               // cur buffer visible to all waves

        // ---- S^T = K_tile * Q^T : four 16x16 key-tiles, K-dim 64 = 2x32 ----
        // sf[jt][r] = score(query = q0+lm, key = 64*ti + 16*jt + 8*lh + r)
        v8f sf[4];
        #pragma unroll
        for (int jt = 0; jt < 4; ++jt) {
            v8f c;
            #pragma unroll
            for (int r = 0; r < 8; ++r) c[r] = 0.0f;
            #pragma unroll
            for (int ch = 0; ch < 2; ++ch) {
                v16h ka;  // A operand: m = lm (key row), Kdim(h) per A pattern
                const _Float16* kr = &sK[cur][jt * 16 + lm][ch * 32];
                #pragma unroll
                for (int e = 0; e < 16; ++e) {
                    const int v = e >> 1, s = e & 1;
                    const int kk = lh * 16 + v * 2 + s;
                    ka[e] = kr[kk];
                }
                c = __builtin_amdgcn_wmma_f32_16x16x32_f16(
                        false, ka, false, qb[ch], (short)0, c, false, false);
            }
            sf[jt] = c;
        }

        // ---- online softmax: scalar state per lane (one query row) ----
        float rmax = -1e30f;
        #pragma unroll
        for (int jt = 0; jt < 4; ++jt) {
            #pragma unroll
            for (int r = 0; r < 8; ++r) rmax = fmaxf(rmax, sf[jt][r]);
        }
        rmax = fmaxf(rmax, __shfl_xor(rmax, 16, 32));  // other 32 keys
        const float nm    = fmaxf(mreg, rmax);
        const float alpha = fast_exp2(mreg - nm);
        mreg = nm;

        // broadcast alpha into O-layout row ownership (q = r + 8*lh)
        float alphaO[8];
        #pragma unroll
        for (int r = 0; r < 8; ++r) alphaO[r] = __shfl(alpha, r + 8 * lh, 32);

        float lsum = 0.0f;
        #pragma unroll
        for (int jt = 0; jt < 4; ++jt) {
            #pragma unroll
            for (int r = 0; r < 8; ++r) {
                sf[jt][r] = fast_exp2(sf[jt][r] - nm);
                lsum += sf[jt][r];
            }
        }
        lreg = lreg * alpha + lsum;

        // ---- P fragments: S^T C-layout == P A-layout, register-only remap ----
        v16h pa[2];
        #pragma unroll
        for (int ch = 0; ch < 2; ++ch) {
            #pragma unroll
            for (int e = 0; e < 16; ++e) {
                const int v = e >> 1, s = e & 1;
                const int g = (v >= 4) ? 1 : 0;
                pa[ch][e] = (_Float16)sf[2 * ch + g][(v & 3) * 2 + s];
            }
        }

        // ---- rescale O, then O += P * V (K-dim 64 = 2 x 32) ----
        #pragma unroll
        for (int t = 0; t < 4; ++t) {
            #pragma unroll
            for (int r = 0; r < 8; ++r) oacc[t][r] *= alphaO[r];
        }
        #pragma unroll
        for (int t = 0; t < 4; ++t) {
            v8f c = oacc[t];
            #pragma unroll
            for (int ch = 0; ch < 2; ++ch) {
                v16h vb;  // B operand: N = lm (h col), Kdim(key) = ch*32+(e>=8?16:0)+lh*8+(e&7)
                const _Float16* vr = &sVt[cur][t * 16 + lm][ch * 32];
                #pragma unroll
                for (int e = 0; e < 16; ++e) vb[e] = vr[((e >= 8) ? 16 : 0) + lh * 8 + (e & 7)];
                c = __builtin_amdgcn_wmma_f32_16x16x32_f16(
                        false, pa[ch], false, vb, (short)0, c, false, false);
            }
            oacc[t] = c;
        }

        // fill the other buffer; safe without a barrier (nobody reads it
        // until after the next __syncthreads)
        if (more) store_tile(cur ^ 1);
        cur ^= 1;
    }

    // ---- epilogue: combine half sums, normalize, store fp32 output ----
    lreg += __shfl_xor(lreg, 16, 32);
    const float linv = 1.0f / lreg;
    float linvO[8];
    #pragma unroll
    for (int r = 0; r < 8; ++r) linvO[r] = __shfl(linv, r + 8 * lh, 32);

    #pragma unroll
    for (int t = 0; t < 4; ++t) {
        #pragma unroll
        for (int r = 0; r < 8; ++r) {
            const int qq = q0 + r + 8 * lh;
            Out[((size_t)batch * kSQ + qq) * (kNH * kHN) + head * kHN + t * 16 + lm] =
                oacc[t][r] * linvO[r];
        }
    }
}

extern "C" void kernel_launch(void* const* d_in, const int* in_sizes, int n_in,
                              void* d_out, int out_size, void* d_ws, size_t ws_size,
                              hipStream_t stream) {
    (void)in_sizes; (void)n_in; (void)out_size; (void)d_ws; (void)ws_size;
    const float* Q = (const float*)d_in[0];
    const float* K = (const float*)d_in[1];
    const float* V = (const float*)d_in[2];
    float* O = (float*)d_out;

    const int grid = (kSQ / kQT) * kB * kNH;  // 16 * 64 = 1024 workgroups
    hipLaunchKernelGGL(fattn_fwd_f16wmma, dim3(grid), dim3(256), 0, stream,
                       Q, K, V, O);
}
